// SOMLayer_68212670595904
// MI455X (gfx1250) — compile-verified, matrix operand order checked by reading, not run on hardware
//
#include <hip/hip_runtime.h>

#define LATENT  256
#define NODES   256
#define ESTRIDE 264                 // 256 + 8 bf16 pad -> 132 dwords, conflict-free b128 LDS reads
#define EPSF    1.1920929e-7f

typedef __attribute__((ext_vector_type(8)))  float  v8f;
typedef __attribute__((ext_vector_type(8)))  __bf16 v8bf;
typedef __attribute__((ext_vector_type(16))) __bf16 v16bf;

__global__ __launch_bounds__(256)
void som_fused_kernel(const float* __restrict__ z, const float* __restrict__ emb,
                      const int* __restrict__ alpha_p,
                      float* __restrict__ out_zq, float* __restrict__ out_zqn,
                      float* __restrict__ out_q,  float* __restrict__ out_bmu) {
    extern __shared__ char smem[];
    __bf16* ehi  = (__bf16*)smem;                       // [256][ESTRIDE] bf16 hi split
    __bf16* elo  = ehi + NODES * ESTRIDE;               // [256][ESTRIDE] bf16 lo split
    float*  e2s  = (float*)(elo + NODES * ESTRIDE);     // [256] node squared norms
    int*    bmus = (int*)(e2s + NODES);                 // [8 waves][16 rows]

    const int tid = threadIdx.x;

    // ---- Stage embeddings to LDS as bf16 hi/lo (3-term split), compute e^2 ----
    {
        const float* er = emb + (size_t)tid * LATENT;
        __bf16* rh = ehi + tid * ESTRIDE;
        __bf16* rl = elo + tid * ESTRIDE;
        float s = 0.f;
        #pragma unroll 4
        for (int k = 0; k < LATENT; k += 4) {
            float4 v = *(const float4*)(er + k);
            __bf16 h0 = (__bf16)v.x, h1 = (__bf16)v.y, h2 = (__bf16)v.z, h3 = (__bf16)v.w;
            rh[k+0] = h0; rh[k+1] = h1; rh[k+2] = h2; rh[k+3] = h3;
            rl[k+0] = (__bf16)(v.x - (float)h0);
            rl[k+1] = (__bf16)(v.y - (float)h1);
            rl[k+2] = (__bf16)(v.z - (float)h2);
            rl[k+3] = (__bf16)(v.w - (float)h3);
            s += v.x*v.x + v.y*v.y + v.z*v.z + v.w*v.w;
        }
        e2s[tid] = s;
    }
    __syncthreads();

    const int lane   = tid & 31;
    const int wave   = tid >> 5;
    const int lhalf  = lane & 15;
    const bool hihalf = (lane & 16) != 0;
    const int m0 = blockIdx.x * 128 + wave * 16;        // 16 rows per wave, 128 per block

    // ---- per-row ||z||^2 (full-wave tree reduce) ----
    float z2r[16];
    #pragma unroll
    for (int r = 0; r < 16; ++r) {
        const float* zr = z + (size_t)(m0 + r) * LATENT + lane * 8;
        float4 a = *(const float4*)zr;
        float4 b = *(const float4*)(zr + 4);
        float s = a.x*a.x + a.y*a.y + a.z*a.z + a.w*a.w
                + b.x*b.x + b.y*b.y + b.z*b.z + b.w*b.w;
        s += __shfl_xor(s, 16); s += __shfl_xor(s, 8); s += __shfl_xor(s, 4);
        s += __shfl_xor(s, 2);  s += __shfl_xor(s, 1);
        z2r[r] = s;
    }

    // ---- GEMM: dot(z_row, e_node) via bf16 WMMA, 3-term split for ~fp32 accuracy ----
    v8f acc[16];
    {
        v8f zv = {0.f,0.f,0.f,0.f,0.f,0.f,0.f,0.f};
        #pragma unroll
        for (int nt = 0; nt < 16; ++nt) acc[nt] = zv;
    }
    const int koff = hihalf ? 8 : 0;                    // ISA 16-bit A/B lane->K mapping
    for (int kc = 0; kc < LATENT; kc += 32) {
        const float* za = z + (size_t)(m0 + lhalf) * LATENT + kc + koff;
        float4 p0 = *(const float4*)(za);
        float4 p1 = *(const float4*)(za + 4);
        float4 p2 = *(const float4*)(za + 16);
        float4 p3 = *(const float4*)(za + 20);
        float av[16];
        av[0]=p0.x; av[1]=p0.y; av[2]=p0.z; av[3]=p0.w;
        av[4]=p1.x; av[5]=p1.y; av[6]=p1.z; av[7]=p1.w;
        av[8]=p2.x; av[9]=p2.y; av[10]=p2.z; av[11]=p2.w;
        av[12]=p3.x; av[13]=p3.y; av[14]=p3.z; av[15]=p3.w;
        v16bf ahi, alo;
        #pragma unroll
        for (int i = 0; i < 16; ++i) {
            __bf16 h = (__bf16)av[i];
            ahi[i] = h;
            alo[i] = (__bf16)(av[i] - (float)h);
        }
        #pragma unroll
        for (int nt = 0; nt < 16; ++nt) {
            const __bf16* bh = ehi + (nt*16 + lhalf) * ESTRIDE + kc + koff;
            const __bf16* bl = elo + (nt*16 + lhalf) * ESTRIDE + kc + koff;
            v8bf b0 = *(const v8bf*)bh;
            v8bf b1 = *(const v8bf*)(bh + 16);
            v8bf c0 = *(const v8bf*)bl;
            v8bf c1 = *(const v8bf*)(bl + 16);
            v16bf bhi = __builtin_shufflevector(b0, b1, 0,1,2,3,4,5,6,7,8,9,10,11,12,13,14,15);
            v16bf blo = __builtin_shufflevector(c0, c1, 0,1,2,3,4,5,6,7,8,9,10,11,12,13,14,15);
            acc[nt] = __builtin_amdgcn_wmma_f32_16x16x32_bf16(false, ahi, false, bhi,
                                                              (short)0, acc[nt], false, false);
            acc[nt] = __builtin_amdgcn_wmma_f32_16x16x32_bf16(false, ahi, false, blo,
                                                              (short)0, acc[nt], false, false);
            acc[nt] = __builtin_amdgcn_wmma_f32_16x16x32_bf16(false, alo, false, bhi,
                                                              (short)0, acc[nt], false, false);
        }
    }

    // ---- epilogue: d = z2 + e2 - 2*dot; argmin; q = t^((a+1)/2); row reductions ----
    const float alpha = (float)(*alpha_p);
    const float invA  = 1.0f / alpha;
    const float ehalf = (alpha + 1.0f) * 0.5f;

    float e2n[16];
    #pragma unroll
    for (int nt = 0; nt < 16; ++nt) e2n[nt] = e2s[nt*16 + lhalf];

    float qsumv[8];
    int   nminv[8];
    #pragma unroll
    for (int v = 0; v < 8; ++v) {
        const float z2 = hihalf ? z2r[v + 8] : z2r[v];   // C layout: lanes16-31 hold rows M=v+8
        float dm = 3.4028235e38f; int nm = 0; float qs = 0.f;
        #pragma unroll
        for (int nt = 0; nt < 16; ++nt) {
            float d = z2 + e2n[nt] - 2.0f * acc[nt][v];
            d = fmaxf(d, 0.0f);
            const int n = nt*16 + lhalf;
            if (d < dm) { dm = d; nm = n; }              // n increases with nt -> first-min kept
            const float t  = 1.0f / (1.0f + d * invA);
            const float qv = exp2f(ehalf * log2f(t));
            qs += qv;
            acc[nt][v] = qv;                             // reuse accumulators for q storage
        }
        #pragma unroll
        for (int m = 8; m >= 1; m >>= 1) {               // reduce across the 16-lane half
            float od = __shfl_xor(dm, m);
            int   on = __shfl_xor(nm, m);
            if (od < dm || (od == dm && on < nm)) { dm = od; nm = on; }
            qs += __shfl_xor(qs, m);
        }
        qsumv[v] = qs;
        nminv[v] = nm;
    }

    // ---- q output: normalized + eps ----
    #pragma unroll
    for (int v = 0; v < 8; ++v) {
        const int row = m0 + v + (hihalf ? 8 : 0);
        const float inv = 1.0f / qsumv[v];
        float* qrow = out_q + (size_t)row * NODES + lhalf;
        #pragma unroll
        for (int nt = 0; nt < 16; ++nt)
            qrow[nt*16] = acc[nt][v] * inv + EPSF;
    }

    // ---- bmu: broadcast to all lanes, park in LDS for the gather loop ----
    int wb[16];
    #pragma unroll
    for (int r = 0; r < 16; ++r)
        wb[r] = __shfl(nminv[r & 7], (r < 8) ? 0 : 16);
    if (lane == 0) {
        #pragma unroll
        for (int r = 0; r < 16; ++r) {
            bmus[wave*16 + r] = wb[r];
            out_bmu[m0 + r] = (float)wb[r];
        }
    }
    __syncthreads();

    // ---- gathers: z_q row + toroidal [self, up, down, right, left] neighbors ----
    for (int r = 0; r < 16; ++r) {
        const int idx = bmus[wave*16 + r];
        const int k1 = idx >> 4, k2 = idx & 15;
        const int nb0 = idx;
        const int nb1 = (((k1 + 15) & 15) << 4) | k2;          // up
        const int nb2 = (((k1 + 1)  & 15) << 4) | k2;          // down
        const int nb3 = (k1 << 4) | ((k2 + 1)  & 15);          // right
        const int nb4 = (k1 << 4) | ((k2 + 15) & 15);          // left
        const int nb[5] = { nb0, nb1, nb2, nb3, nb4 };
        const size_t row = (size_t)(m0 + r);
        {
            const float4* s = (const float4*)(emb + (size_t)idx * LATENT);
            float4* d4 = (float4*)(out_zq + row * LATENT);
            d4[lane*2]     = s[lane*2];
            d4[lane*2 + 1] = s[lane*2 + 1];
        }
        #pragma unroll
        for (int j = 0; j < 5; ++j) {
            const float4* s = (const float4*)(emb + (size_t)nb[j] * LATENT);
            float4* d4 = (float4*)(out_zqn + (row * 5 + j) * LATENT);
            d4[lane*2]     = s[lane*2];
            d4[lane*2 + 1] = s[lane*2 + 1];
        }
    }
}

extern "C" void kernel_launch(void* const* d_in, const int* in_sizes, int n_in,
                              void* d_out, int out_size, void* d_ws, size_t ws_size,
                              hipStream_t stream) {
    const float* z   = (const float*)d_in[0];
    const float* emb = (const float*)d_in[1];
    const int*   a   = (const int*)d_in[2];

    const size_t nrows = (size_t)in_sizes[0] / LATENT;   // 65536
    float* out     = (float*)d_out;
    float* out_zq  = out;                                // [N,256]
    float* out_zqn = out + nrows * 256;                  // [N,5,256]
    float* out_q   = out + nrows * 1536;                 // [N,256]
    float* out_bmu = out + nrows * 1792;                 // [N] (as float)

    const size_t shmem = (size_t)NODES * ESTRIDE * 2 * sizeof(__bf16)  // e hi/lo
                       + NODES * sizeof(float)                          // e^2
                       + 8 * 16 * sizeof(int);                          // bmu staging
    const int blocks = (int)(nrows / 128);               // 128 rows / block (8 waves x 16)
    som_fused_kernel<<<blocks, 256, shmem, stream>>>(z, emb, a, out_zq, out_zqn, out_q, out_bmu);
}
